// DeformableFeatureAggregation_77335181132311
// MI455X (gfx1250) — matrix operand order repaired and checked by reading, not compiled, not run on hardware
//
#include <hip/hip_runtime.h>
#include <hip/hip_bf16.h>

typedef __attribute__((ext_vector_type(16))) _Float16 v16h;
typedef __attribute__((ext_vector_type(8)))  _Float16 v8h;
typedef __attribute__((ext_vector_type(8)))  float    v8f;

#define N_ANCHOR 900
#define NPAD     912          // 57 tiles of 16
#define P_SAMPLE 8
#define EMBED    256
#define GROUPS   8
#define GD       32
#define LEVELS   4
#define CAMS     6
#define NW1      1536         // GROUPS*CAMS*LEVELS*P
#define CLP      192          // CAMS*LEVELS*P
#define EPS_F    1e-5f

// ---------------------------------------------------------------------------
// feat = instance_feature + anchor_embed, converted to f16, padded to NPAD rows
// ---------------------------------------------------------------------------
__global__ void make_featA(const float* __restrict__ inst,
                           const float* __restrict__ aemb,
                           _Float16* __restrict__ outA) {
  int idx = blockIdx.x * blockDim.x + threadIdx.x;       // [0, NPAD*256)
  if (idx >= NPAD * EMBED) return;
  int m = idx >> 8;
  float v = 0.f;
  if (m < N_ANCHOR) v = inst[idx] + aemb[idx];
  outA[idx] = (_Float16)v;
}

// ---------------------------------------------------------------------------
// Pack a row-major f32 weight W[K=256][Nw] into the wave32 WMMA-B layout:
// Bp[((tn*8 + ks)*32 + lane)*16 + j]:
//   lane<16 : B[ks*32 + (j<8 ? j : 8+j)][tn*16 + lane]      (j = 0..15)
//   lane>=16: B[ks*32 + 8 + (j<8 ? j : 8+j)][tn*16 + lane - 16]
// -> each lane of the GEMM does one contiguous 32-byte load per K-step.
// ---------------------------------------------------------------------------
__global__ void pack_B(const float* __restrict__ W, _Float16* __restrict__ Bp,
                       int Nw, int ntilesN) {
  int idx = blockIdx.x * blockDim.x + threadIdx.x;
  int total = ntilesN * 8 * 32 * 16;
  if (idx >= total) return;
  int j    = idx & 15;
  int lane = (idx >> 4) & 31;
  int ks   = (idx >> 9) & 7;
  int tn   = idx >> 12;
  int n = tn * 16 + (lane & 15);
  int k = ks * 32 + ((lane & 16) ? 8 : 0) + j + ((j & 8) ? 8 : 0);
  Bp[idx] = (_Float16)W[k * Nw + n];
}

// ---------------------------------------------------------------------------
// Keypoint projection: per (n, cam, p) -> normalized pixel coords + mask
// ---------------------------------------------------------------------------
__global__ void project_pts(const float* __restrict__ anchor,   // [900][16]
                            const float* __restrict__ pm,       // [6][4][4]
                            const float* __restrict__ wh,       // [6][2]
                            float* __restrict__ px, float* __restrict__ py,
                            int* __restrict__ mk) {
  int idx = blockIdx.x * blockDim.x + threadIdx.x;   // n*48 + cam*8 + p
  if (idx >= N_ANCHOR * CAMS * P_SAMPLE) return;
  int p   = idx & 7;
  int cam = (idx >> 3) % CAMS;
  int n   = idx / (CAMS * P_SAMPLE);
  float ax = anchor[n * 16 + p * 2 + 0];
  float ay = anchor[n * 16 + p * 2 + 1];
  const float* M = pm + cam * 16;                    // row-major 4x4
  float X = M[0] * ax + M[1] * ay + M[3];            // z = 0, w = 1
  float Y = M[4] * ax + M[5] * ay + M[7];
  float Z = M[8] * ax + M[9] * ay + M[11];
  float denom = fmaxf(Z, EPS_F);
  float xp = X / denom, yp = Y / denom;
  bool m = (Z > EPS_F) && (xp > 0.f) && (yp > 0.f);
  float xn = xp / wh[cam * 2 + 0];
  float yn = yp / wh[cam * 2 + 1];
  m = m && (xn < 1.f) && (yn < 1.f);
  int o = (n * CAMS + cam) * P_SAMPLE + p;
  px[o] = xn; py[o] = yn; mk[o] = m ? 1 : 0;
}

__global__ void any_cam_k(const int* __restrict__ mk, int* __restrict__ anyb) {
  int idx = blockIdx.x * blockDim.x + threadIdx.x;   // n*8 + p
  if (idx >= N_ANCHOR * P_SAMPLE) return;
  int n = idx >> 3, p = idx & 7;
  int a = 0;
#pragma unroll
  for (int cam = 0; cam < CAMS; ++cam) a |= mk[(n * CAMS + cam) * P_SAMPLE + p];
  anyb[idx] = a;
}

// ---------------------------------------------------------------------------
// WMMA GEMM: Out[M][Nw] = A[Mpad][256](f16) x B(packed f16) + bias (+resid)
// One wave per 16x16 output tile, K=256 in 8 steps of v_wmma_f32_16x16x32_f16.
// ---------------------------------------------------------------------------
__global__ void gemm_wmma(const _Float16* __restrict__ A,
                          const _Float16* __restrict__ Bp,
                          const float* __restrict__ bias,
                          const float* __restrict__ resid,   // may be null
                          float* __restrict__ Out,
                          int Mreal, int Nw, int ntilesN, int totalTiles) {
  int wave = threadIdx.x >> 5;
  int lane = threadIdx.x & 31;
  int tile = blockIdx.x * (blockDim.x >> 5) + wave;
  if (tile >= totalTiles) return;                    // wave-uniform exit
  int tm = tile / ntilesN;
  int tn = tile - tm * ntilesN;

  // A row for this lane: lanes 0-15 carry K-chunks {0..7,16..23},
  // lanes 16-31 carry {8..15,24..31} of the same 16 M rows.
  const _Float16* arow = A + (size_t)(tm * 16 + (lane & 15)) * EMBED
                           + ((lane & 16) ? 8 : 0);
  const _Float16* bbase = Bp + ((size_t)tn * 8) * (32 * 16) + lane * 16;

  v8f c = {0.f, 0.f, 0.f, 0.f, 0.f, 0.f, 0.f, 0.f};
#pragma unroll
  for (int ks = 0; ks < 8; ++ks) {
    v8h alo = *(const v8h*)(arow + ks * 32);
    v8h ahi = *(const v8h*)(arow + ks * 32 + 16);
    v16h a = __builtin_shufflevector(alo, ahi, 0, 1, 2, 3, 4, 5, 6, 7,
                                     8, 9, 10, 11, 12, 13, 14, 15);
    v16h b = *(const v16h*)(bbase + ks * (32 * 16));
    c = __builtin_amdgcn_wmma_f32_16x16x32_f16(false, a, false, b,
                                               (short)0, c, false, false);
  }

  int ncol  = tn * 16 + (lane & 15);
  int mbase = tm * 16 + ((lane & 16) ? 8 : 0);
  float bv = bias[ncol];
#pragma unroll
  for (int r = 0; r < 8; ++r) {
    int m = mbase + r;
    if (m < Mreal) {
      float v = c[r] + bv;
      if (resid) v += resid[(size_t)m * Nw + ncol];
      Out[(size_t)m * Nw + ncol] = v;
    }
  }
}

// ---------------------------------------------------------------------------
// Masked softmax over the 192-long (cam, level, p) axis, per (anchor, group).
// In-place on W[900][1536]; col layout ((cam*4 + l)*8 + p)*8 + g.
// ---------------------------------------------------------------------------
__global__ void softmax_w(float* __restrict__ W, const int* __restrict__ mk,
                          const int* __restrict__ anyb) {
  int idx = blockIdx.x * blockDim.x + threadIdx.x;   // n*8 + g
  if (idx >= N_ANCHOR * GROUPS) return;
  int n = idx >> 3, g = idx & 7;
  float* row = W + (size_t)n * NW1;
  float mx = -3.0e38f;
  for (int i = 0; i < CLP; ++i) {
    int cam = i >> 5, p = i & 7;
    bool masked = (!mk[(n * CAMS + cam) * P_SAMPLE + p]) &&
                  anyb[n * P_SAMPLE + p];
    if (!masked) mx = fmaxf(mx, row[i * GROUPS + g]);
  }
  float s = 0.f;
  for (int i = 0; i < CLP; ++i) {
    int cam = i >> 5, p = i & 7;
    bool masked = (!mk[(n * CAMS + cam) * P_SAMPLE + p]) &&
                  anyb[n * P_SAMPLE + p];
    if (!masked) s += expf(row[i * GROUPS + g] - mx);
  }
  float inv = 1.f / s;
  for (int i = 0; i < CLP; ++i) {
    int cam = i >> 5, p = i & 7;
    bool masked = (!mk[(n * CAMS + cam) * P_SAMPLE + p]) &&
                  anyb[n * P_SAMPLE + p];
    float e = masked ? 0.f : expf(row[i * GROUPS + g] - mx) * inv;
    row[i * GROUPS + g] = e;
  }
}

__global__ void zero_pad_feats(_Float16* __restrict__ feats) {
  int i = blockIdx.x * blockDim.x + threadIdx.x;
  if (i < (NPAD - N_ANCHOR) * EMBED)
    feats[N_ANCHOR * EMBED + i] = (_Float16)0.f;
}

// ---------------------------------------------------------------------------
// Bilinear sampling + group-weighted aggregation.
// One block per anchor; thread = channel c (group g = c/32).
// Weights + coords staged in LDS; skip flag is block-uniform per (cam,p).
// ---------------------------------------------------------------------------
__global__ void sample_agg(const float* __restrict__ fm0,
                           const float* __restrict__ fm1,
                           const float* __restrict__ fm2,
                           const float* __restrict__ fm3,
                           const float* __restrict__ Wsm,
                           const float* __restrict__ px,
                           const float* __restrict__ py,
                           const int* __restrict__ mk,
                           const int* __restrict__ anyb,
                           _Float16* __restrict__ feats) {
  __shared__ float ws[NW1];
  __shared__ float pxs[CAMS * P_SAMPLE], pys[CAMS * P_SAMPLE];
  __shared__ int   skip[CAMS * P_SAMPLE];
  int n = blockIdx.x;
  int c = threadIdx.x;                                  // 0..255
  for (int i = c; i < NW1; i += EMBED) ws[i] = Wsm[(size_t)n * NW1 + i];
  if (c < CAMS * P_SAMPLE) {
    int o = n * CAMS * P_SAMPLE + c;
    pxs[c] = px[o]; pys[c] = py[o];
    skip[c] = (!mk[o]) && anyb[n * P_SAMPLE + (c & 7)];
  }
  __syncthreads();

  int g = c >> 5;
  const float* fms[LEVELS] = {fm0, fm1, fm2, fm3};
  const int Hs[LEVELS] = {64, 32, 16, 8};
  const int Wd_[LEVELS] = {176, 88, 44, 22};
  float acc = 0.f;
  for (int cam = 0; cam < CAMS; ++cam) {
    for (int p = 0; p < P_SAMPLE; ++p) {
      int cp = cam * P_SAMPLE + p;
      if (skip[cp]) continue;                           // block-uniform
      float xn = pxs[cp], yn = pys[cp];
#pragma unroll
      for (int l = 0; l < LEVELS; ++l) {
        int H = Hs[l], Wd = Wd_[l];
        float w = ws[((cam * LEVELS + l) * P_SAMPLE + p) * GROUPS + g];
        const float* base = fms[l] + (size_t)(cam * EMBED + c) * (H * Wd);
        float fx = xn * (float)Wd - 0.5f;
        float fy = yn * (float)H - 0.5f;
        float x0f = floorf(fx), y0f = floorf(fy);
        float wx = fx - x0f, wy = fy - y0f;
        int x0 = (int)x0f, y0 = (int)y0f;
        bool xok0 = (x0 >= 0) && (x0 < Wd);
        bool xok1 = (x0 + 1 >= 0) && (x0 + 1 < Wd);
        bool yok0 = (y0 >= 0) && (y0 < H);
        bool yok1 = (y0 + 1 >= 0) && (y0 + 1 < H);
        float v00 = 0.f, v10 = 0.f, v01 = 0.f, v11 = 0.f;
        if (yok0) {
          const float* r0 = base + (size_t)y0 * Wd;
          if (xok0) v00 = r0[x0];
          if (xok1) v10 = r0[x0 + 1];
        }
        if (yok1) {
          const float* r1 = base + (size_t)(y0 + 1) * Wd;
          if (xok0) v01 = r1[x0];
          if (xok1) v11 = r1[x0 + 1];
        }
        float s = (v00 * (1.f - wx) + v10 * wx) * (1.f - wy) +
                  (v01 * (1.f - wx) + v11 * wx) * wy;
        acc += s * w;
      }
    }
  }
  feats[(size_t)n * EMBED + c] = (_Float16)acc;
}

// ---------------------------------------------------------------------------
static inline size_t align256(size_t x) { return (x + 255) & ~(size_t)255; }

extern "C" void kernel_launch(void* const* d_in, const int* in_sizes, int n_in,
                              void* d_out, int out_size, void* d_ws, size_t ws_size,
                              hipStream_t stream) {
  const float* inst  = (const float*)d_in[0];
  const float* anch  = (const float*)d_in[1];
  const float* aemb  = (const float*)d_in[2];
  const float* fm0   = (const float*)d_in[3];
  const float* fm1   = (const float*)d_in[4];
  const float* fm2   = (const float*)d_in[5];
  const float* fm3   = (const float*)d_in[6];
  const float* pm    = (const float*)d_in[7];
  const float* wh    = (const float*)d_in[8];
  const float* wfc_w = (const float*)d_in[9];
  const float* wfc_b = (const float*)d_in[10];
  const float* out_w = (const float*)d_in[11];
  const float* out_b = (const float*)d_in[12];
  float* out = (float*)d_out;

  // workspace carve-up
  char* ws = (char*)d_ws;
  size_t off = 0;
  _Float16* featA = (_Float16*)(ws + off); off = align256(off + (size_t)NPAD * EMBED * 2);
  _Float16* Bp1   = (_Float16*)(ws + off); off = align256(off + (size_t)96 * 8 * 32 * 16 * 2);
  _Float16* Bp2   = (_Float16*)(ws + off); off = align256(off + (size_t)16 * 8 * 32 * 16 * 2);
  float*    Wraw  = (float*)   (ws + off); off = align256(off + (size_t)NPAD * NW1 * 4);
  float*    pxb   = (float*)   (ws + off); off = align256(off + (size_t)N_ANCHOR * CAMS * P_SAMPLE * 4);
  float*    pyb   = (float*)   (ws + off); off = align256(off + (size_t)N_ANCHOR * CAMS * P_SAMPLE * 4);
  int*      mkb   = (int*)     (ws + off); off = align256(off + (size_t)N_ANCHOR * CAMS * P_SAMPLE * 4);
  int*      anyb  = (int*)     (ws + off); off = align256(off + (size_t)N_ANCHOR * P_SAMPLE * 4);
  _Float16* featsH= (_Float16*)(ws + off); off = align256(off + (size_t)NPAD * EMBED * 2);
  (void)ws_size; (void)in_sizes; (void)n_in; (void)out_size;

  const int T = 256;
  // 1) f16 A = inst + anchor_embed (zero-padded to 912 rows)
  make_featA<<<(NPAD * EMBED + T - 1) / T, T, 0, stream>>>(inst, aemb, featA);
  // 2) pack weight matrices into WMMA-B tile layout
  pack_B<<<(96 * 8 * 32 * 16 + T - 1) / T, T, 0, stream>>>(wfc_w, Bp1, NW1, 96);
  pack_B<<<(16 * 8 * 32 * 16 + T - 1) / T, T, 0, stream>>>(out_w, Bp2, EMBED, 16);
  // 3) projection + visibility
  project_pts<<<(N_ANCHOR * CAMS * P_SAMPLE + T - 1) / T, T, 0, stream>>>(
      anch, pm, wh, pxb, pyb, mkb);
  any_cam_k<<<(N_ANCHOR * P_SAMPLE + T - 1) / T, T, 0, stream>>>(mkb, anyb);
  // 4) GEMM1: Wraw = feat @ wfc_w + wfc_b   (57x96 = 5472 tiles, 8 waves/block)
  {
    int tiles = (NPAD / 16) * 96;
    gemm_wmma<<<(tiles + 7) / 8, T, 0, stream>>>(featA, Bp1, wfc_b, nullptr,
                                                 Wraw, N_ANCHOR, NW1, 96, tiles);
  }
  // 5) masked softmax over (cam, level, p), per (anchor, group), in place
  softmax_w<<<(N_ANCHOR * GROUPS + T - 1) / T, T, 0, stream>>>(Wraw, mkb, anyb);
  // 6) bilinear sample + aggregate -> feats (f16, padded)
  zero_pad_feats<<<((NPAD - N_ANCHOR) * EMBED + T - 1) / T, T, 0, stream>>>(featsH);
  sample_agg<<<N_ANCHOR, T, 0, stream>>>(fm0, fm1, fm2, fm3, Wraw, pxb, pyb,
                                         mkb, anyb, featsH);
  // 7) GEMM2: out = feats @ out_w + out_b + inst   (57x16 = 912 tiles)
  {
    int tiles = (NPAD / 16) * 16;
    gemm_wmma<<<(tiles + 7) / 8, T, 0, stream>>>(featsH, Bp2, out_b, inst,
                                                 out, N_ANCHOR, EMBED, 16, tiles);
  }
}
